// GT32dim_3_modes_6_dec_linear_81286551044467
// MI455X (gfx1250) — matrix operand order repaired and checked by reading, not compiled, not run on hardware
//
#include <hip/hip_runtime.h>

#define C 32
#define D_CAT 288
#define NB 64
#define NLAYERS 9

typedef __attribute__((ext_vector_type(16))) _Float16 v16h;
typedef __attribute__((ext_vector_type(8)))  float    v8f;

// ---- monotonic float<->uint encoding for atomic max on floats ----
__device__ __forceinline__ unsigned f2ord(float f) {
  unsigned u = __float_as_uint(f);
  return (u & 0x80000000u) ? ~u : (u | 0x80000000u);
}
__device__ __forceinline__ float ord2f(unsigned u) {
  u = (u & 0x80000000u) ? (u & 0x7FFFFFFFu) : ~u;
  return __uint_as_float(u);
}

// Load one A-fragment row (16x32 f16 layout) with float4 (b128) loads.
// lane<16 -> K 0..15 ; lane>=16 -> K 16..31  (kb = hi*8, start = 2*kb)
__device__ __forceinline__ v16h load_a_frag(const float* __restrict__ row, int kb) {
  float4 s0 = *(const float4*)(row + 2 * kb);
  float4 s1 = *(const float4*)(row + 2 * kb + 4);
  float4 s2 = *(const float4*)(row + 2 * kb + 8);
  float4 s3 = *(const float4*)(row + 2 * kb + 12);
  v16h a;
  a[0]  = (_Float16)s0.x; a[1]  = (_Float16)s0.y; a[2]  = (_Float16)s0.z; a[3]  = (_Float16)s0.w;
  a[4]  = (_Float16)s1.x; a[5]  = (_Float16)s1.y; a[6]  = (_Float16)s1.z; a[7]  = (_Float16)s1.w;
  a[8]  = (_Float16)s2.x; a[9]  = (_Float16)s2.y; a[10] = (_Float16)s2.z; a[11] = (_Float16)s2.w;
  a[12] = (_Float16)s3.x; a[13] = (_Float16)s3.y; a[14] = (_Float16)s3.z; a[15] = (_Float16)s3.w;
  return a;
}

// ---------------- per-layer projections: q,k,v,skip via WMMA ----------------
// Each wave handles a 32-row slab (two 16x16 M-tiles), reusing the 8 B-fragments.
__global__ __launch_bounds__(256) void proj_wmma_kernel(
    const float* __restrict__ xs, int ss,
    const float* __restrict__ xd, int sd,
    const float* __restrict__ Wq, const float* __restrict__ bq,
    const float* __restrict__ Wk, const float* __restrict__ bk,
    const float* __restrict__ Wv, const float* __restrict__ bv,
    const float* __restrict__ Ws, const float* __restrict__ bs,
    float* __restrict__ outq, float* __restrict__ outk,
    float* __restrict__ outv, float* __restrict__ outs, int n)
{
  // Transposed f16 weights: wl[m*1024 + ncol*32 + krow] = W_m[krow*32 + ncol].
  // B-fragment (col, khalf) is then 16 CONTIGUOUS halves -> 2x ds_load_b128.
  __shared__ __attribute__((aligned(32))) _Float16 wl[4 * C * C];
  for (int idx = threadIdx.x; idx < 4 * C * C; idx += 256) {
    int m = idx >> 10;
    int t = idx & (C * C - 1);
    int ncol = t >> 5, krow = t & 31;
    const float* w = (m == 0) ? Wq : (m == 1) ? Wk : (m == 2) ? Wv : Ws;
    wl[idx] = (_Float16)w[krow * C + ncol];
  }
  __syncthreads();

  const int wave = threadIdx.x >> 5;   // wave32
  const int lane = threadIdx.x & 31;
  const int lo16 = lane & 15;
  const int hi   = lane >> 4;
  const int row0 = (blockIdx.x * 8 + wave) * 32;
  if (row0 >= n) return;
  const bool has2 = (row0 + 16) < n;   // n is a multiple of 16

  const int kb = hi * 8;
  const int r0 = row0 + lo16;
  v16h aS0 = load_a_frag(xs + (size_t)r0 * ss, kb);
  v16h aD0 = load_a_frag(xd + (size_t)r0 * sd, kb);
  v16h aS1, aD1;
  if (has2) {
    const int r1 = r0 + 16;
    aS1 = load_a_frag(xs + (size_t)r1 * ss, kb);
    aD1 = load_a_frag(xd + (size_t)r1 * sd, kb);
  }

  const int krow0 = hi * 16;           // B: lanes 0-15 K=0..15, lanes 16-31 K=16..31
  #pragma unroll
  for (int ct = 0; ct < 2; ++ct) {
    const int col = ct * 16 + lo16;
    const int boff = col * C + krow0;  // contiguous 16 halves, 32B aligned
    v16h Bq = *(const v16h*)(wl + 0 * C * C + boff);
    v16h Bk = *(const v16h*)(wl + 1 * C * C + boff);
    v16h Bv = *(const v16h*)(wl + 2 * C * C + boff);
    v16h Bs = *(const v16h*)(wl + 3 * C * C + boff);
    const float vbq = bq[col], vbk = bk[col], vbv = bv[col], vbs = bs[col];

    #pragma unroll
    for (int mt = 0; mt < 2; ++mt) {
      if (mt == 1 && !has2) break;     // wave-uniform: EXEC stays all-ones
      v16h aS = (mt == 0) ? aS0 : aS1;
      v16h aD = (mt == 0) ? aD0 : aD1;
      v8f aq, ak, av, ask;
      #pragma unroll
      for (int j = 0; j < 8; ++j) { aq[j] = vbq; ak[j] = vbk; av[j] = vbv; ask[j] = vbs; }
      aq  = __builtin_amdgcn_wmma_f32_16x16x32_f16(false, aD, false, Bq, (short)0, aq,  false, false);
      ak  = __builtin_amdgcn_wmma_f32_16x16x32_f16(false, aS, false, Bk, (short)0, ak,  false, false);
      av  = __builtin_amdgcn_wmma_f32_16x16x32_f16(false, aS, false, Bv, (short)0, av,  false, false);
      ask = __builtin_amdgcn_wmma_f32_16x16x32_f16(false, aD, false, Bs, (short)0, ask, false, false);
      const int rbase = row0 + mt * 16 + hi * 8;
      #pragma unroll
      for (int j = 0; j < 8; ++j) {    // D: VGPR j -> M=j (lanes 0-15) / M=8+j (lanes 16-31)
        const size_t oo = (size_t)(rbase + j) * C + col;
        outq[oo] = aq[j]; outk[oo] = ak[j]; outv[oo] = av[j]; outs[oo] = ask[j];
      }
    }
  }
}

// ---------------- edge-phase kernels ----------------
__global__ void wecol_kernel(const float* __restrict__ We, float* __restrict__ wecol) {
  int t = threadIdx.x;
  if (t < 3 * C) {
    int rel = t >> 5, c = t & 31;
    float s = 0.f;
    for (int i = 0; i < C; ++i) s += We[rel * C * C + i * C + c];
    wecol[t] = s;                      // colsum(We): edge_feat @ We == ea * wecol
  }
}

__global__ void edge_init_kernel(unsigned* __restrict__ menc, float* __restrict__ denom,
                                 float* __restrict__ agg, int n) {
  int t = blockIdx.x * blockDim.x + threadIdx.x;
  if (t >= n * C) return;
  agg[t] = 0.f;
  if ((t & 31) == 0) { int nd = t >> 5; menc[nd] = 0x007FFFFFu; denom[nd] = 0.f; }  // enc(-inf)
}

__global__ void edge_logits_kernel(const int* __restrict__ ei, int E_,
                                   const float* __restrict__ qm, const float* __restrict__ km,
                                   const float* __restrict__ ea, const float* __restrict__ wec,
                                   float* __restrict__ logits, unsigned* __restrict__ menc)
{
  int e = blockIdx.x * blockDim.x + threadIdx.x;
  if (e >= E_) return;
  int src = ei[e], dst = ei[E_ + e];
  const float4* qp = (const float4*)(qm + (size_t)dst * C);
  const float4* kp = (const float4*)(km + (size_t)src * C);
  float a = wec ? ea[e] : 0.f;
  float dot = 0.f;
  #pragma unroll
  for (int c4 = 0; c4 < C / 4; ++c4) {
    float4 q4 = qp[c4];
    float4 k4 = kp[c4];
    if (wec) {
      float4 w4 = ((const float4*)wec)[c4];
      k4.x += a * w4.x; k4.y += a * w4.y; k4.z += a * w4.z; k4.w += a * w4.w;
    }
    dot += q4.x * k4.x + q4.y * k4.y + q4.z * k4.z + q4.w * k4.w;
  }
  float lg = dot * 0.17677669529663687f;   // 1/sqrt(32)
  logits[e] = lg;
  atomicMax(&menc[dst], f2ord(lg));
}

// Per (edge, 8-channel quarter): ex = exp(logit - m[dst]); denom += ex; agg += ex*(v+e)
__global__ void edge_accum_kernel(const int* __restrict__ ei, int E_,
                                  const float* __restrict__ vm,
                                  const float* __restrict__ ea, const float* __restrict__ wec,
                                  const float* __restrict__ logits, const unsigned* __restrict__ menc,
                                  float* __restrict__ denom, float* __restrict__ agg)
{
  int t = blockIdx.x * blockDim.x + threadIdx.x;
  if (t >= E_ * 4) return;
  int e = t >> 2, qt = t & 3;
  int src = ei[e], dst = ei[E_ + e];
  float ex = __expf(logits[e] - ord2f(menc[dst]));
  if (qt == 0) atomicAdd(&denom[dst], ex);
  float a = wec ? ea[e] : 0.f;
  float4 v0 = ((const float4*)(vm + (size_t)src * C + qt * 8))[0];
  float4 v1 = ((const float4*)(vm + (size_t)src * C + qt * 8))[1];
  float vv[8] = {v0.x, v0.y, v0.z, v0.w, v1.x, v1.y, v1.z, v1.w};
  float* ap = agg + (size_t)dst * C + qt * 8;
  #pragma unroll
  for (int c = 0; c < 8; ++c) {
    float x = vv[c];
    if (wec) x += a * wec[qt * 8 + c];
    atomicAdd(&ap[c], ex * x);
  }
}

__global__ void finalize_kernel(const float* __restrict__ agg, const float* __restrict__ denom,
                                const float* __restrict__ skm,
                                float* __restrict__ xcur, float* __restrict__ states,
                                int layer, int n)
{
  int t = blockIdx.x * blockDim.x + threadIdx.x;
  if (t >= n * C) return;
  int node = t >> 5, c = t & 31;
  float val = tanhf(agg[t] / (denom[node] + 1e-16f) + skm[t]);
  xcur[t] = val;
  states[(size_t)node * D_CAT + layer * C + c] = val;
}

// ---------------- pooling (mean commutes with lin1) ----------------
__global__ void pool_init_kernel(float* __restrict__ ps, float* __restrict__ cnt) {
  int t = blockIdx.x * blockDim.x + threadIdx.x;
  if (t < NB * D_CAT) ps[t] = 0.f;
  if (t < NB) cnt[t] = 0.f;
}

__global__ void pool_sum_kernel(const float* __restrict__ states, const int* __restrict__ batch,
                                float* __restrict__ ps, float* __restrict__ cnt, int n) {
  int t = blockIdx.x * blockDim.x + threadIdx.x;
  if (t >= n * D_CAT) return;
  int node = t / D_CAT, d = t - node * D_CAT;
  int b = batch[node];
  atomicAdd(&ps[b * D_CAT + d], states[t]);
  if (d == 0) atomicAdd(&cnt[b], 1.f);
}

__global__ void pool_mean_kernel(float* __restrict__ ps, const float* __restrict__ cnt) {
  int t = blockIdx.x * blockDim.x + threadIdx.x;
  if (t >= NB * D_CAT) return;
  ps[t] /= fmaxf(cnt[t / D_CAT], 1.f);
}

// ---------------- lin1 on pooled [64,288] @ [288,288] via WMMA ----------------
__global__ __launch_bounds__(32) void lin1_wmma_kernel(
    const float* __restrict__ ps, const float* __restrict__ W1,
    const float* __restrict__ b1, float* __restrict__ out)
{
  const int lane = threadIdx.x & 31;
  const int lo16 = lane & 15;
  const int hi   = lane >> 4;
  const int nt = blockIdx.x;           // 0..17
  const int mt = blockIdx.y;           // 0..3
  const int col = nt * 16 + lo16;
  const int r   = mt * 16 + lo16;

  v8f acc;
  const float bb = b1[col];
  #pragma unroll
  for (int j = 0; j < 8; ++j) acc[j] = bb;

  const float* pr = ps + (size_t)r * D_CAT;
  for (int kk = 0; kk < D_CAT / 32; ++kk) {
    v16h a = load_a_frag(pr + kk * 32, hi * 8);
    v16h b;
    const int krow = kk * 32 + hi * 16;
    #pragma unroll
    for (int e = 0; e < 16; ++e)
      b[e] = (_Float16)W1[(size_t)(krow + e) * D_CAT + col];
    acc = __builtin_amdgcn_wmma_f32_16x16x32_f16(false, a, false, b, (short)0, acc, false, false);
  }
  #pragma unroll
  for (int j = 0; j < 8; ++j)
    out[(size_t)(mt * 16 + j + hi * 8) * D_CAT + col] = acc[j];
}

// ---------------- tiny MLP head ----------------
__global__ void head_linear_kernel(const float* __restrict__ in, const float* __restrict__ W,
                                   const float* __restrict__ b, float* __restrict__ out,
                                   int K, int Nout, int relu) {
  int t = blockIdx.x * blockDim.x + threadIdx.x;
  if (t >= NB * Nout) return;
  int r = t / Nout, c = t - r * Nout;
  float s = b[c];
  for (int i = 0; i < K; ++i) s += in[r * K + i] * W[i * Nout + c];
  out[t] = relu ? fmaxf(s, 0.f) : s;
}

__global__ void logsoftmax_kernel(const float* __restrict__ h, float* __restrict__ out) {
  int r = blockIdx.x * blockDim.x + threadIdx.x;
  if (r >= NB) return;
  float mx = -INFINITY;
  for (int c = 0; c < 10; ++c) mx = fmaxf(mx, h[r * 10 + c]);
  float s = 0.f;
  for (int c = 0; c < 10; ++c) s += __expf(h[r * 10 + c] - mx);
  float lse = mx + __logf(s);
  for (int c = 0; c < 10; ++c) out[r * 10 + c] = h[r * 10 + c] - lse;
}

// ---------------- host orchestration ----------------
extern "C" void kernel_launch(void* const* d_in, const int* in_sizes, int n_in,
                              void* d_out, int out_size, void* d_ws, size_t ws_size,
                              hipStream_t stream) {
  (void)n_in; (void)out_size; (void)ws_size;
  const float* inter_emb = (const float*)d_in[0];
  const float* edge_attr = (const float*)d_in[1];
  const float* uv_emb    = (const float*)d_in[2];
  const int*   ei_inter  = (const int*)d_in[3];
  const int*   ei_uvt    = (const int*)d_in[4];
  const int*   ei_tuv    = (const int*)d_in[5];
  const int*   batch     = (const int*)d_in[6];
  const float* Wq  = (const float*)d_in[7];
  const float* bqp = (const float*)d_in[8];
  const float* Wk  = (const float*)d_in[9];
  const float* bkp = (const float*)d_in[10];
  const float* Wv  = (const float*)d_in[11];
  const float* bvp = (const float*)d_in[12];
  const float* Wsk = (const float*)d_in[13];
  const float* bsp = (const float*)d_in[14];
  const float* We  = (const float*)d_in[15];
  const float* W1 = (const float*)d_in[16]; const float* b1 = (const float*)d_in[17];
  const float* W2 = (const float*)d_in[18]; const float* b2 = (const float*)d_in[19];
  const float* W3 = (const float*)d_in[20]; const float* b3 = (const float*)d_in[21];
  const float* W4 = (const float*)d_in[22]; const float* b4 = (const float*)d_in[23];
  const float* W5 = (const float*)d_in[24]; const float* b5 = (const float*)d_in[25];
  const float* W6 = (const float*)d_in[26]; const float* b6 = (const float*)d_in[27];

  const int N = in_sizes[2] / C;     // uv_target_emb is [N, C]
  const int E = in_sizes[1];         // edge_attr is [E]

  float* f = (float*)d_ws;
  size_t o = 0;
  float* qb     = f + o; o += (size_t)N * C;
  float* kb     = f + o; o += (size_t)N * C;
  float* vb     = f + o; o += (size_t)N * C;
  float* skb    = f + o; o += (size_t)N * C;
  float* xcur   = f + o; o += (size_t)N * C;
  float* states = f + o; o += (size_t)N * D_CAT;
  float* logits = f + o; o += (size_t)E;
  unsigned* menc = (unsigned*)(f + o); o += (size_t)N;
  float* denom  = f + o; o += (size_t)N;
  float* agg    = f + o; o += (size_t)N * C;
  float* wecol  = f + o; o += 3 * C;
  float* cnt    = f + o; o += NB;
  float* ps     = f + o; o += NB * D_CAT;
  float* pooled = f + o; o += NB * D_CAT;
  float* h2 = f + o; o += NB * 144;
  float* h3 = f + o; o += NB * 144;
  float* h4 = f + o; o += NB * 72;
  float* h5 = f + o; o += NB * 72;
  float* h6 = f + o; o += NB * 10;

  wecol_kernel<<<1, 128, 0, stream>>>(We, wecol);

  const int projBlocks = ((N + 255) / 256);          // 32 rows per wave, 8 waves per block
  const int nc = N * C;
  for (int i = 0; i < NLAYERS; ++i) {
    const float *xs, *xd; int ss, sd;
    const int* ei; const float* wec = nullptr;
    if (i % 3 == 0) {
      if (i == 0) { xs = inter_emb; ss = 2 * C; xd = inter_emb + C; sd = 2 * C; } // f0, f1
      else        { xs = xcur; ss = C; xd = xcur; sd = C; }
      ei = ei_inter; wec = wecol + (i / 3) * C;
    } else if (i % 3 == 1) {
      xs = xcur; ss = C; xd = uv_emb; sd = C; ei = ei_uvt;
    } else {
      xs = uv_emb; ss = C; xd = xcur; sd = C; ei = ei_tuv;
    }
    proj_wmma_kernel<<<projBlocks, 256, 0, stream>>>(
        xs, ss, xd, sd,
        Wq  + (size_t)i * C * C, bqp + i * C,
        Wk  + (size_t)i * C * C, bkp + i * C,
        Wv  + (size_t)i * C * C, bvp + i * C,
        Wsk + (size_t)i * C * C, bsp + i * C,
        qb, kb, vb, skb, N);
    edge_init_kernel<<<(nc + 255) / 256, 256, 0, stream>>>(menc, denom, agg, N);
    edge_logits_kernel<<<(E + 255) / 256, 256, 0, stream>>>(ei, E, qb, kb, edge_attr, wec, logits, menc);
    edge_accum_kernel<<<(E * 4 + 255) / 256, 256, 0, stream>>>(ei, E, vb, edge_attr, wec, logits, menc, denom, agg);
    finalize_kernel<<<(nc + 255) / 256, 256, 0, stream>>>(agg, denom, skb, xcur, states, i, N);
  }

  pool_init_kernel<<<(NB * D_CAT + 255) / 256, 256, 0, stream>>>(ps, cnt);
  {
    int total = N * D_CAT;
    pool_sum_kernel<<<(total + 255) / 256, 256, 0, stream>>>(states, batch, ps, cnt, N);
  }
  pool_mean_kernel<<<(NB * D_CAT + 255) / 256, 256, 0, stream>>>(ps, cnt);
  lin1_wmma_kernel<<<dim3(D_CAT / 16, NB / 16), 32, 0, stream>>>(ps, W1, b1, pooled);
  head_linear_kernel<<<(NB * 144 + 255) / 256, 256, 0, stream>>>(pooled, W2, b2, h2, D_CAT, 144, 1);
  head_linear_kernel<<<(NB * 144 + 255) / 256, 256, 0, stream>>>(h2, W3, b3, h3, 144, 144, 0);
  head_linear_kernel<<<(NB * 72 + 255) / 256, 256, 0, stream>>>(h3, W4, b4, h4, 144, 72, 0);
  head_linear_kernel<<<(NB * 72 + 255) / 256, 256, 0, stream>>>(h4, W5, b5, h5, 72, 72, 0);
  head_linear_kernel<<<(NB * 10 + 255) / 256, 256, 0, stream>>>(h5, W6, b6, h6, 72, 10, 0);
  logsoftmax_kernel<<<1, 64, 0, stream>>>(h6, (float*)d_out);
}
